// DeepSpeedSelfAttention_2783138807823
// MI455X (gfx1250) — compile-verified, matrix operand order checked by reading, not run on hardware
//
#include <hip/hip_runtime.h>
#include <hip/hip_bf16.h>

// ---------------------------------------------------------------------------
// DeepSpeed-style self-attention for MI455X (gfx1250, wave32, WMMA).
// B=2, S=2048, H=2048, HEADS=16, D=128. All GEMMs in bf16 WMMA (f32 accum),
// attention done flash-style (scores never hit HBM). GEMM staging uses
// double-buffered LDS fed by GLOBAL_LOAD_ASYNC_TO_LDS (ASYNCcnt-tracked).
// ---------------------------------------------------------------------------

#define HID   2048
#define SEQ   2048
#define NB    2
#define NHEAD 16
#define HD    128
#define MTOT  (NB * SEQ)        // 4096

typedef __attribute__((ext_vector_type(16))) __bf16          v16bf;
typedef __attribute__((ext_vector_type(16))) unsigned short  v16u;
typedef __attribute__((ext_vector_type(8)))  unsigned short  v8u;
typedef __attribute__((ext_vector_type(8)))  float           v8f;
typedef __attribute__((ext_vector_type(4)))  int             v4i;

// ---- async global -> LDS copy (CDNA5), with safe fallback -----------------
#if defined(__has_builtin) && __has_builtin(__builtin_amdgcn_global_load_async_to_lds_b128)
#define HAVE_ASYNC_LDS 1
#define ASYNC_CP16(g, l)                                                     \
    __builtin_amdgcn_global_load_async_to_lds_b128(                          \
        (__attribute__((address_space(1))) v4i*)(g),                         \
        (__attribute__((address_space(3))) v4i*)(l), 0, 0)
#if __has_builtin(__builtin_amdgcn_s_wait_asynccnt)
#define WAIT_ASYNC() __builtin_amdgcn_s_wait_asynccnt(0)
#else
#define WAIT_ASYNC() asm volatile("s_wait_asynccnt 0" ::: "memory")
#endif
#else
#define HAVE_ASYNC_LDS 0
#define ASYNC_CP16(g, l) (*(uint4*)(l) = *(const uint4*)(g))
#define WAIT_ASYNC() ((void)0)
#endif

__device__ __forceinline__ unsigned short f2bf(float f) {
    unsigned int u = __builtin_bit_cast(unsigned int, f);
    u = u + 0x7fffu + ((u >> 16) & 1u);          // round-to-nearest-even
    return (unsigned short)(u >> 16);
}
__device__ __forceinline__ float bf2f(unsigned short u) {
    unsigned int v = ((unsigned int)u) << 16;
    return __builtin_bit_cast(float, v);
}

// Load one 16x32 bf16 A/B^T fragment (ISA 7.12.2 layout): lane holds row
// (lane&15); K-halves {0..7,16..23} for lanes 0-15, {8..15,24..31} for 16-31.
// p0 must already point at row_base + (lane<16 ? 0 : 8); p0 and p0+16 are the
// two contiguous 8-half (16B) chunks -> two ds_load_b128 / global_load_b128.
__device__ __forceinline__ v16bf load_frag(const unsigned short* p0) {
    v8u lo = *(const v8u*)p0;
    v8u hi = *(const v8u*)(p0 + 16);
    v16u r;
#pragma unroll
    for (int i = 0; i < 8; ++i) { r[i] = lo[i]; r[i + 8] = hi[i]; }
    return __builtin_bit_cast(v16bf, r);
}

__device__ __forceinline__ v8f wmma_bf16(v16bf a, v16bf b, v8f c) {
    return __builtin_amdgcn_wmma_f32_16x16x32_bf16(false, a, false, b,
                                                   (short)0, c, false, false);
}

// ------------------------------ RMSNorm ------------------------------------
__global__ __launch_bounds__(256) void rmsnorm_kernel(const float* __restrict__ x,
                                                      const float* __restrict__ w,
                                                      unsigned short* __restrict__ h) {
    __shared__ float red[256];
    const int row = blockIdx.x;
    const float* xr = x + (size_t)row * HID;
    float xv[8];
    float ss = 0.f;
#pragma unroll
    for (int i = 0; i < 8; ++i) {
        xv[i] = xr[threadIdx.x + i * 256];
        ss += xv[i] * xv[i];
    }
    red[threadIdx.x] = ss;
    __syncthreads();
    for (int off = 128; off > 0; off >>= 1) {
        if (threadIdx.x < off) red[threadIdx.x] += red[threadIdx.x + off];
        __syncthreads();
    }
    const float rstd = rsqrtf(red[0] * (1.0f / HID) + 1e-6f);
#pragma unroll
    for (int i = 0; i < 8; ++i) {
        const int d = threadIdx.x + i * 256;
        h[(size_t)row * HID + d] = f2bf(w[d] * xv[i] * rstd);
    }
}

// --------------------------- f32 -> bf16 convert ---------------------------
__global__ __launch_bounds__(256) void cvt_kernel(const float* __restrict__ src,
                                                  unsigned short* __restrict__ dst,
                                                  int n) {
    for (int i = blockIdx.x * 256 + threadIdx.x; i < n; i += gridDim.x * 256)
        dst[i] = f2bf(src[i]);
}

// ------------------------------- RoPE --------------------------------------
// In-place on bf16 [B, HEADS, S, D]; one thread per (d, d+64) pair.
__global__ __launch_bounds__(256) void rope_kernel(unsigned short* __restrict__ p,
                                                   const int* __restrict__ pos) {
    const int idx = blockIdx.x * 256 + threadIdx.x;   // 2*16*2048*64 total
    const int j  = idx & 63;
    const int s  = (idx >> 6) & (SEQ - 1);
    const int bh = idx >> 17;
    const int b  = bh >> 4;
    const float t   = (float)pos[b * SEQ + s];
    const float inv = __powf(10000.0f, -(float)j * (1.0f / 64.0f));
    float sn, c;
    __sincosf(t * inv, &sn, &c);
    const size_t e1 = (size_t)bh * SEQ * HD + (size_t)s * HD + j;
    const size_t e2 = e1 + 64;
    const float x1 = bf2f(p[e1]);
    const float x2 = bf2f(p[e2]);
    p[e1] = f2bf(x1 * c - x2 * sn);
    p[e2] = f2bf(x2 * c + x1 * sn);
}

// --------------------------- WMMA GEMM (y = A * W^T) -----------------------
// A: [M=4096, 2048] bf16 row-major.  W: [2048(out), 2048(in)] bf16 row-major
// (so the GEMM B-matrix transpose == W rows: contiguous copies into LDS).
// Block tile 128x128, 8 waves each own 32x64, K-step 32. Double-buffered LDS
// with async global->LDS copies: next tile streams in while WMMA consumes the
// current one; one barrier per K-step.
// MODE 0: scatter bf16 into [B, HEADS, S, D];  MODE 1: f32 [M, 2048].
template <int MODE>
__global__ __launch_bounds__(256) void gemm_wmma(const unsigned short* __restrict__ A,
                                                 const unsigned short* __restrict__ W,
                                                 void* __restrict__ out) {
    __shared__ __align__(16) unsigned short As[2][128][40];   // +8 pad: conflict-free b128
    __shared__ __align__(16) unsigned short Bs[2][128][40];

    const int tid  = threadIdx.x;
    const int wave = tid >> 5;
    const int lane = tid & 31;
    const int l16  = lane & 15;
    const int lhf  = (lane & 16) ? 8 : 0;
    const int wm   = wave >> 1;        // 0..3
    const int wn   = wave & 1;         // 0..1
    const int m0   = blockIdx.y * 128;
    const int n0   = blockIdx.x * 128;

    // each thread stages 2 x 16B chunks per tile
    const int srow0 = tid >> 2,         sch0 = (tid & 3) << 3;
    const int srow1 = (tid + 256) >> 2, sch1 = (tid & 3) << 3;

    v8f acc[2][4];
#pragma unroll
    for (int mi = 0; mi < 2; ++mi)
#pragma unroll
        for (int ni = 0; ni < 4; ++ni) acc[mi][ni] = (v8f){};

    // prologue: stage k0 = 0 into buffer 0
    ASYNC_CP16(A + (size_t)(m0 + srow0) * HID + sch0, &As[0][srow0][sch0]);
    ASYNC_CP16(A + (size_t)(m0 + srow1) * HID + sch1, &As[0][srow1][sch1]);
    ASYNC_CP16(W + (size_t)(n0 + srow0) * HID + sch0, &Bs[0][srow0][sch0]);
    ASYNC_CP16(W + (size_t)(n0 + srow1) * HID + sch1, &Bs[0][srow1][sch1]);

    int buf = 0;
    for (int k0 = 0; k0 < HID; k0 += 32) {
        WAIT_ASYNC();        // our copies into `buf` have landed in LDS
        __syncthreads();     // everyone's copies have landed; prev reads done

        if (k0 + 32 < HID) { // stream next tile while we compute this one
            const int nb = buf ^ 1;
            const int kn = k0 + 32;
            ASYNC_CP16(A + (size_t)(m0 + srow0) * HID + kn + sch0, &As[nb][srow0][sch0]);
            ASYNC_CP16(A + (size_t)(m0 + srow1) * HID + kn + sch1, &As[nb][srow1][sch1]);
            ASYNC_CP16(W + (size_t)(n0 + srow0) * HID + kn + sch0, &Bs[nb][srow0][sch0]);
            ASYNC_CP16(W + (size_t)(n0 + srow1) * HID + kn + sch1, &Bs[nb][srow1][sch1]);
        }

        v16bf af[2], bf_[4];
#pragma unroll
        for (int mi = 0; mi < 2; ++mi)
            af[mi] = load_frag(&As[buf][wm * 32 + mi * 16 + l16][lhf]);
#pragma unroll
        for (int ni = 0; ni < 4; ++ni)
            bf_[ni] = load_frag(&Bs[buf][wn * 64 + ni * 16 + l16][lhf]);
#pragma unroll
        for (int mi = 0; mi < 2; ++mi)
#pragma unroll
            for (int ni = 0; ni < 4; ++ni)
                acc[mi][ni] = wmma_bf16(af[mi], bf_[ni], acc[mi][ni]);

        buf ^= 1;
    }

    // Epilogue (C layout: VGPR r -> row r / r+8 per lane-half, col = lane&15)
#pragma unroll
    for (int mi = 0; mi < 2; ++mi)
#pragma unroll
        for (int ni = 0; ni < 4; ++ni)
#pragma unroll
            for (int r = 0; r < 8; ++r) {
                const int m = m0 + wm * 32 + mi * 16 + r + lhf;
                const int n = n0 + wn * 64 + ni * 16 + l16;
                const float v = acc[mi][ni][r];
                if (MODE == 0) {
                    const int head = n >> 7, d = n & 127;
                    const int b = m >> 11, s = m & (SEQ - 1);
                    ((unsigned short*)out)[(((size_t)(b * NHEAD + head)) * SEQ + s) * HD + d] = f2bf(v);
                } else {
                    ((float*)out)[(size_t)m * HID + n] = v;
                }
            }
}

// --------------------------- Flash attention -------------------------------
// grid = (S/128, B*HEADS). 8 waves, each owns 16 q rows; Q frags live in
// VGPRs; K tile row-major in LDS (async copies), V tile transposed in LDS;
// P goes through a per-wave LDS buffer to convert WMMA C-layout -> A-layout.
__global__ __launch_bounds__(256) void attn_kernel(const unsigned short* __restrict__ Q,
                                                   const unsigned short* __restrict__ K,
                                                   const unsigned short* __restrict__ V,
                                                   unsigned short* __restrict__ Ctx) {
    __shared__ __align__(16) unsigned short Kt[64][136];     // [kk][d], +8 pad
    __shared__ __align__(16) unsigned short Vt[128][72];     // [d][kk], +8 pad
    __shared__ __align__(16) unsigned short Pl[8][16][72];   // per-wave P

    const int qt   = blockIdx.x;
    const int bh   = blockIdx.y;
    const int b    = bh >> 4;
    const int head = bh & 15;
    const int tid  = threadIdx.x;
    const int wave = tid >> 5;
    const int lane = tid & 31;
    const int l16  = lane & 15;
    const int lhf  = (lane & 16) ? 8 : 0;

    const size_t hoff = (size_t)bh * SEQ * HD;
    const unsigned short* Qh = Q + hoff;
    const unsigned short* Kh = K + hoff;
    const unsigned short* Vh = V + hoff;

    const int q0 = qt * 128 + wave * 16;

    v16bf aq[4];
#pragma unroll
    for (int g = 0; g < 4; ++g)
        aq[g] = load_frag(Qh + (size_t)(q0 + l16) * HD + g * 32 + lhf);

    v8f o[8];
    float mrow[8], lrow[8];
#pragma unroll
    for (int t = 0; t < 8; ++t) o[t] = (v8f){};
#pragma unroll
    for (int r = 0; r < 8; ++r) { mrow[r] = -1e30f; lrow[r] = 0.f; }

    const float scale = 0.08838834764831845f;   // 1/sqrt(128)
    const int ktmax = qt * 2 + 1;               // causal: last needed 64-wide k tile

    for (int kt = 0; kt <= ktmax; ++kt) {
        __syncthreads();                        // Kt/Vt free for reuse
#pragma unroll
        for (int i = 0; i < 4; ++i) {           // stage K (async) + V^T (transpose)
            const int idx = tid + i * 256;      // 1024 16B chunks
            const int row = idx >> 4;           // kk 0..63
            const int ch  = (idx & 15) << 3;    // d chunk
            const size_t g = (size_t)(kt * 64 + row) * HD + ch;
            ASYNC_CP16(Kh + g, &Kt[row][ch]);
            unsigned short tmp[8];
            *(uint4*)tmp = *(const uint4*)(Vh + g);
#pragma unroll
            for (int e = 0; e < 8; ++e) Vt[ch + e][row] = tmp[e];
        }
        WAIT_ASYNC();
        __syncthreads();

        // S = Q K^T : 4 n-subtiles x 4 k-steps
        v8f sv[4];
#pragma unroll
        for (int j = 0; j < 4; ++j) {
            v8f s = (v8f){};
#pragma unroll
            for (int g = 0; g < 4; ++g) {
                v16bf bk = load_frag(&Kt[j * 16 + l16][g * 32 + lhf]);
                s = wmma_bf16(aq[g], bk, s);
            }
            sv[j] = s;
        }

        // scale + causal mask
#pragma unroll
        for (int j = 0; j < 4; ++j) {
            const int kkg = kt * 64 + j * 16 + l16;
#pragma unroll
            for (int r = 0; r < 8; ++r) {
                const int qg = q0 + r + lhf;
                float s = sv[j][r] * scale;
                if (kkg > qg) s += -10000.0f;
                sv[j][r] = s;
            }
        }

        // row max (row lives across 16 lanes of one half-wave)
        float rm[8];
#pragma unroll
        for (int r = 0; r < 8; ++r) {
            float v = fmaxf(fmaxf(sv[0][r], sv[1][r]), fmaxf(sv[2][r], sv[3][r]));
            v = fmaxf(v, __shfl_xor(v, 1, 16));
            v = fmaxf(v, __shfl_xor(v, 2, 16));
            v = fmaxf(v, __shfl_xor(v, 4, 16));
            v = fmaxf(v, __shfl_xor(v, 8, 16));
            rm[r] = v;
        }

        float alpha[8];
#pragma unroll
        for (int r = 0; r < 8; ++r) {
            const float mn = fmaxf(mrow[r], rm[r]);
            alpha[r] = __expf(mrow[r] - mn);
            mrow[r] = mn;
        }

        // P = exp(s - m): write bf16 to per-wave LDS, accumulate row sums
        float rs[8] = {0.f, 0.f, 0.f, 0.f, 0.f, 0.f, 0.f, 0.f};
#pragma unroll
        for (int j = 0; j < 4; ++j)
#pragma unroll
            for (int r = 0; r < 8; ++r) {
                const float p = __expf(sv[j][r] - mrow[r]);
                rs[r] += p;
                Pl[wave][r + lhf][j * 16 + l16] = f2bf(p);
            }
#pragma unroll
        for (int r = 0; r < 8; ++r) {
            float v = rs[r];
            v += __shfl_xor(v, 1, 16);
            v += __shfl_xor(v, 2, 16);
            v += __shfl_xor(v, 4, 16);
            v += __shfl_xor(v, 8, 16);
            lrow[r] = lrow[r] * alpha[r] + v;
        }

#pragma unroll
        for (int t = 0; t < 8; ++t)
#pragma unroll
            for (int r = 0; r < 8; ++r) o[t][r] *= alpha[r];

        asm volatile("" ::: "memory");   // keep P stores before A-frag reloads

        // O += P * V : 2 k-steps x 8 d-subtiles
#pragma unroll
        for (int g = 0; g < 2; ++g) {
            v16bf ap = load_frag(&Pl[wave][l16][g * 32 + lhf]);
#pragma unroll
            for (int t = 0; t < 8; ++t) {
                v16bf bv = load_frag(&Vt[t * 16 + l16][g * 32 + lhf]);
                o[t] = wmma_bf16(ap, bv, o[t]);
            }
        }
    }

    // normalize + write ctx [B, S, H] bf16
    float invl[8];
#pragma unroll
    for (int r = 0; r < 8; ++r) invl[r] = 1.0f / lrow[r];
    const size_t cbase = ((size_t)b * SEQ) * HID + (size_t)head * HD;
#pragma unroll
    for (int t = 0; t < 8; ++t)
#pragma unroll
        for (int r = 0; r < 8; ++r) {
            const int qg = q0 + r + lhf;
            const int d  = t * 16 + l16;
            Ctx[cbase + (size_t)qg * HID + d] = f2bf(o[t][r] * invl[r]);
        }
}

// ------------------------------- launcher ----------------------------------
extern "C" void kernel_launch(void* const* d_in, const int* in_sizes, int n_in,
                              void* d_out, int out_size, void* d_ws, size_t ws_size,
                              hipStream_t stream) {
    (void)in_sizes; (void)n_in; (void)out_size; (void)ws_size;

    const float* x      = (const float*)d_in[0];
    const float* norm_w = (const float*)d_in[1];
    const float* wq     = (const float*)d_in[2];
    const float* wk     = (const float*)d_in[3];
    const float* wv     = (const float*)d_in[4];
    const float* wo     = (const float*)d_in[5];
    const int*   pos    = (const int*)d_in[6];

    char* ws = (char*)d_ws;
    const size_t SZ_ACT = (size_t)MTOT * HID * 2;          // 16 MiB
    const size_t SZ_W   = (size_t)HID * HID * 2;           // 8 MiB
    unsigned short* h_bf   = (unsigned short*)(ws);
    unsigned short* wq_bf  = (unsigned short*)(ws + SZ_ACT);
    unsigned short* wk_bf  = (unsigned short*)(ws + SZ_ACT + SZ_W);
    unsigned short* wv_bf  = (unsigned short*)(ws + SZ_ACT + 2 * SZ_W);
    unsigned short* wo_bf  = (unsigned short*)(ws + SZ_ACT + 3 * SZ_W);
    unsigned short* q_bf   = (unsigned short*)(ws + SZ_ACT + 4 * SZ_W);
    unsigned short* k_bf   = (unsigned short*)(ws + 2 * SZ_ACT + 4 * SZ_W);
    unsigned short* v_bf   = (unsigned short*)(ws + 3 * SZ_ACT + 4 * SZ_W);
    unsigned short* ctx_bf = (unsigned short*)(ws + 4 * SZ_ACT + 4 * SZ_W);

    rmsnorm_kernel<<<MTOT, 256, 0, stream>>>(x, norm_w, h_bf);

    const int wN = HID * HID;
    cvt_kernel<<<2048, 256, 0, stream>>>(wq, wq_bf, wN);
    cvt_kernel<<<2048, 256, 0, stream>>>(wk, wk_bf, wN);
    cvt_kernel<<<2048, 256, 0, stream>>>(wv, wv_bf, wN);
    cvt_kernel<<<2048, 256, 0, stream>>>(wo, wo_bf, wN);

    dim3 ggrid(HID / 128, MTOT / 128);   // (16, 32)
    gemm_wmma<0><<<ggrid, 256, 0, stream>>>(h_bf, wq_bf, (void*)q_bf);
    gemm_wmma<0><<<ggrid, 256, 0, stream>>>(h_bf, wk_bf, (void*)k_bf);
    gemm_wmma<0><<<ggrid, 256, 0, stream>>>(h_bf, wv_bf, (void*)v_bf);

    const int ropeN = NB * NHEAD * SEQ * 64;               // 4,194,304
    rope_kernel<<<ropeN / 256, 256, 0, stream>>>(q_bf, pos);
    rope_kernel<<<ropeN / 256, 256, 0, stream>>>(k_bf, pos);

    dim3 agrid(SEQ / 128, NB * NHEAD);   // (16, 32)
    attn_kernel<<<agrid, 256, 0, stream>>>(q_bf, k_bf, v_bf, ctx_bf);

    gemm_wmma<1><<<ggrid, 256, 0, stream>>>(ctx_bf, wo_bf, d_out);
}
